// AttentionBlock_70755291234598
// MI455X (gfx1250) — matrix operand order repaired and checked by reading, not compiled, not run on hardware
//
#include <hip/hip_runtime.h>
#include <hip/hip_bf16.h>

// ---------------------------------------------------------------------------
// Problem constants (B, C, L, HEADS, GROUPS from the reference)
// ---------------------------------------------------------------------------
constexpr int kB = 4;
constexpr int kC = 512;
constexpr int kL = 2048;
constexpr int kH = 8;
constexpr int kD = 64;          // C / HEADS
constexpr int kG = 8;           // groups
constexpr float kEps   = 1e-5f;
constexpr float kScale = 0.044194173824159216f;  // 512^-0.5 (full C, per reference)

typedef __attribute__((ext_vector_type(16))) _Float16 v16h;
typedef __attribute__((ext_vector_type(8)))  _Float16 v8h;
typedef __attribute__((ext_vector_type(8)))  float    v8f;
typedef __attribute__((ext_vector_type(4)))  unsigned int v4u;
typedef __attribute__((ext_vector_type(4)))  int v4i;
typedef __attribute__((ext_vector_type(8)))  int v8i;

#if defined(__AMDGCN__) && __has_builtin(__builtin_amdgcn_tensor_load_to_lds) && \
    __has_builtin(__builtin_amdgcn_s_wait_tensorcnt)
#define USE_TDM 1
#else
#define USE_TDM 0
#endif

// ---------------------------------------------------------------------------
// WMMA fragment loaders (wave32, 16x16x32 f16)
//   A 16x32: lane holds row m=lane&15, 16 contiguous K at (lane>>4)*16
//   B 32x16: lane holds column n=lane&15, 16 contiguous K at (lane>>4)*16
// ---------------------------------------------------------------------------
__device__ inline v16h load_a16(const _Float16* row_ptr, int kbase, int lane) {
  return *(const v16h*)(row_ptr + kbase + ((lane >> 4) << 4));
}

__device__ inline v8f wmma_f16(v16h a, v16h b, v8f c) {
  return __builtin_amdgcn_wmma_f32_16x16x32_f16(false, a, false, b,
                                                (short)0, c, false, false);
}

#if USE_TDM
// ---------------------------------------------------------------------------
// Tensor Data Mover: build D# (groups 0-1 per ISA 8.3/8.4, data_size = 8B)
// and issue TENSOR_LOAD_TO_LDS. 2-D tile: tile_d0 x tile_d1 elements of 8B,
// rows separated by stride0 (8B units). All args must be wave-uniform.
// ---------------------------------------------------------------------------
__device__ inline void tdm_load_to_lds(unsigned lds_off, const void* gptr,
                                       unsigned tensor_d0, unsigned tensor_d1,
                                       unsigned tile_d0, unsigned tile_d1,
                                       unsigned long long stride0) {
  const unsigned long long ga = (unsigned long long)(uintptr_t)gptr;
  v4u g0;
  g0.x = 1u;                                          // count=1, user load
  g0.y = lds_off;                                     // LDS byte address
  g0.z = (unsigned)ga;                                // global_addr[31:0]
  g0.w = (unsigned)((ga >> 32) & 0x1FFFFFFull) | (2u << 30);  // [56:32] | type=2
  v8i g1;
  g1[0] = (int)(3u << 16);                            // data_size = 8B
  g1[1] = (int)((tensor_d0 & 0xFFFFu) << 16);         // tensor_dim0[15:0]
  g1[2] = (int)((tensor_d0 >> 16) | ((tensor_d1 & 0xFFFFu) << 16));
  g1[3] = (int)((tensor_d1 >> 16) | (tile_d0 << 16)); // tensor_dim1 hi | tile_dim0
  g1[4] = (int)(tile_d1 & 0xFFFFu);                   // tile_dim1 (tile_dim2 = 0)
  g1[5] = (int)(unsigned)(stride0 & 0xFFFFFFFFull);   // tensor_dim0_stride lo
  g1[6] = (int)(unsigned)((stride0 >> 32) & 0xFFFFull);
  g1[7] = 0;
  v4i gz = {0, 0, 0, 0};
#if __clang_major__ >= 23
  v8i gz8 = {0, 0, 0, 0, 0, 0, 0, 0};
  __builtin_amdgcn_tensor_load_to_lds(g0, g1, gz, gz, gz8, 0);
#else
  __builtin_amdgcn_tensor_load_to_lds(g0, g1, gz, gz, 0);
#endif
}
#endif  // USE_TDM

// ---------------------------------------------------------------------------
// Kernel 1: cast qkv_w / out_w to f16
// ---------------------------------------------------------------------------
__global__ void cast_weights_kernel(const float* __restrict__ qkv_w,
                                    const float* __restrict__ out_w,
                                    _Float16* __restrict__ wq,
                                    _Float16* __restrict__ wo) {
  int i = blockIdx.x * blockDim.x + threadIdx.x;
  if (i < 3 * kC * kC) wq[i] = (_Float16)qkv_w[i];
  if (i < kC * kC)     wo[i] = (_Float16)out_w[i];
}

// ---------------------------------------------------------------------------
// Kernel 2: GroupNorm statistics, one block per (batch, group)
// ---------------------------------------------------------------------------
__global__ void gn_stats_kernel(const float* __restrict__ x,
                                float* __restrict__ stats) {
  __shared__ float s_sum[256];
  __shared__ float s_sq[256];
  const int bg = blockIdx.x;                       // b*8+g, contiguous slab
  const float* p = x + (size_t)bg * (kC / kG) * kL;
  const int n = (kC / kG) * kL;                    // 131072
  float sum = 0.f, sq = 0.f;
  for (int i = threadIdx.x; i < n; i += 256) {
    float v = p[i];
    sum += v; sq += v * v;
  }
  s_sum[threadIdx.x] = sum; s_sq[threadIdx.x] = sq;
  __syncthreads();
  for (int s = 128; s > 0; s >>= 1) {
    if (threadIdx.x < s) {
      s_sum[threadIdx.x] += s_sum[threadIdx.x + s];
      s_sq[threadIdx.x]  += s_sq[threadIdx.x + s];
    }
    __syncthreads();
  }
  if (threadIdx.x == 0) {
    float mean = s_sum[0] / (float)n;
    float var  = s_sq[0] / (float)n - mean * mean;
    stats[bg * 2]     = mean;
    stats[bg * 2 + 1] = rsqrtf(var + kEps);
  }
}

// ---------------------------------------------------------------------------
// Kernel 3: apply GN + LDS tile transpose to [B][L][C] f16.
// ---------------------------------------------------------------------------
__global__ __launch_bounds__(256)
void gn_apply_kernel(const float* __restrict__ x,
                     const float* __restrict__ gn_w,
                     const float* __restrict__ gn_b,
                     const float* __restrict__ stats,
                     _Float16* __restrict__ hnT) {
  __shared__ _Float16 tile[32][72];                // [c_local][l_local], padded
  const int b  = blockIdx.z;
  const int c0 = blockIdx.y * 32;
  const int l0 = blockIdx.x * 64;
  const int t  = threadIdx.x;
  {
    const int cl = t >> 3;                         // 0..31
    const int ll = (t & 7) * 8;                    // 0,8,...,56
    const int c  = c0 + cl;
    const int bg = b * kG + (c >> 6);
    const float mean = stats[bg * 2], rstd = stats[bg * 2 + 1];
    const float w = gn_w[c], bias = gn_b[c];
    const float* xp = x + ((size_t)(b * kC + c)) * kL + l0 + ll;
#pragma unroll
    for (int i = 0; i < 8; ++i)
      tile[cl][ll + i] = (_Float16)((xp[i] - mean) * rstd * w + bias);
  }
  __syncthreads();
  {
    const int ll = t >> 2;                         // 0..63
    const int cl = (t & 3) * 8;                    // 0,8,16,24
    v8h vv;
#pragma unroll
    for (int i = 0; i < 8; ++i) vv[i] = tile[cl + i][ll];
    *(v8h*)(hnT + ((size_t)(b * kL + l0 + ll)) * kC + c0 + cl) = vv;
  }
}

// ---------------------------------------------------------------------------
// Kernel 4: QKV GEMM  [3C x C] * [C x L] per batch, WMMA f16->f32.
// ---------------------------------------------------------------------------
__global__ __launch_bounds__(128)
void qkv_gemm_kernel(const _Float16* __restrict__ wq,
                     const _Float16* __restrict__ hnT,
                     const float* __restrict__ qkv_b,
                     _Float16* __restrict__ qout,
                     _Float16* __restrict__ kout,
                     _Float16* __restrict__ vT) {
  const int lane = threadIdx.x & 31;
  const int wave = threadIdx.x >> 5;
  const int b  = blockIdx.z;
  const int m0 = blockIdx.x * 64 + (wave & 1) * 32;
  const int n0 = blockIdx.y * 128 + (wave >> 1) * 64;
  const _Float16* Bb = hnT + (size_t)b * kL * kC;

  v8f acc[2][4] = {};
  for (int kc = 0; kc < kC; kc += 32) {
    v16h a[2];
    a[0] = load_a16(wq + (size_t)(m0 + (lane & 15)) * kC, kc, lane);
    a[1] = load_a16(wq + (size_t)(m0 + 16 + (lane & 15)) * kC, kc, lane);
#pragma unroll
    for (int nj = 0; nj < 4; ++nj) {
      const _Float16* bp =
          Bb + (size_t)(n0 + nj * 16 + (lane & 15)) * kC + kc + ((lane >> 4) << 4);
      v16h bf = *(const v16h*)bp;
      acc[0][nj] = wmma_f16(a[0], bf, acc[0][nj]);
      acc[1][nj] = wmma_f16(a[1], bf, acc[1][nj]);
    }
  }
#pragma unroll
  for (int mi = 0; mi < 2; ++mi) {
#pragma unroll
    for (int r = 0; r < 8; ++r) {
      const int o = m0 + mi * 16 + r + ((lane >> 4) << 3);
      const float bias = qkv_b[o];
      const int part = o >> 9;
      const int c = o & (kC - 1);
      const int h = c >> 6, d = c & 63;
      const size_t bh = (size_t)(b * kH + h);
#pragma unroll
      for (int nj = 0; nj < 4; ++nj) {
        const int l = n0 + nj * 16 + (lane & 15);
        _Float16 hv = (_Float16)(acc[mi][nj][r] + bias);
        if (part == 0)      qout[(bh * kL + l) * kD + d] = hv;
        else if (part == 1) kout[(bh * kL + l) * kD + d] = hv;
        else                vT[(bh * kD + d) * kL + l]   = hv;
      }
    }
  }
}

// ---------------------------------------------------------------------------
// Kernel 5: flash attention. WG = 4 waves x 32 Q rows = 128 rows of one (b,h).
//           K/V chunks double-buffered in LDS via the Tensor Data Mover
//           (TENSORcnt-tracked DMA issued by wave 0, overlapped with WMMA);
//           falls back to cooperative vector copies if TDM is unavailable.
// ---------------------------------------------------------------------------
__global__ __launch_bounds__(128)
void attn_kernel(const _Float16* __restrict__ q,
                 const _Float16* __restrict__ k,
                 const _Float16* __restrict__ vT,
                 _Float16* __restrict__ oT) {
  alignas(64) __shared__ _Float16 ldsK[2][32 * 64];  // [lk][d]  2 x 4 KB
  alignas(64) __shared__ _Float16 ldsV[2][64 * 32];  // [d][lk]  2 x 4 KB
  alignas(64) __shared__ _Float16 ldsP[4 * 32 * 32]; // per-wave P, 8 KB
  const int lane = threadIdx.x & 31;
  const int wave = threadIdx.x >> 5;
  const int tid  = threadIdx.x;
  const int b = blockIdx.z, h = blockIdx.y;
  const int lq0 = blockIdx.x * 128 + wave * 32;    // this wave: rows [lq0, lq0+32)
  const _Float16* qb = q  + ((size_t)(b * kH + h) * kL) * kD;
  const _Float16* kb = k  + ((size_t)(b * kH + h) * kL) * kD;
  const _Float16* vb = vT + ((size_t)(b * kH + h) * kD) * kL;
  _Float16* lp = ldsP + wave * (32 * 32);

  v16h aq[2][2];
#pragma unroll
  for (int mi = 0; mi < 2; ++mi) {
    const _Float16* qrow = qb + (size_t)(lq0 + mi * 16 + (lane & 15)) * kD;
    aq[mi][0] = load_a16(qrow, 0, lane);
    aq[mi][1] = load_a16(qrow, 32, lane);
  }

  v8f acc[2][4] = {};
  float rowM[2][8], rowL[2][8];
#pragma unroll
  for (int mi = 0; mi < 2; ++mi)
#pragma unroll
    for (int r = 0; r < 8; ++r) { rowM[mi][r] = -3.0e38f; rowL[mi][r] = 0.f; }

#if USE_TDM
  if (wave == 0) {
    // K chunk: flat 4 KB (32 rows x 64 halves contiguous) -> 1-D tile of 512x8B
    tdm_load_to_lds((unsigned)(uintptr_t)&ldsK[0][0], kb, 512, 1, 512, 1, 512);
    // V chunk: 64 rows x 64B from vT (row stride = 4 KB = 512x8B)
    tdm_load_to_lds((unsigned)(uintptr_t)&ldsV[0][0], vb, 512, 64, 8, 64, 512);
  }
#endif

  for (int i = 0; i < kL / 32; ++i) {
    const int cur = i & 1;
    const int lk0 = i * 32;
#if USE_TDM
    if (wave == 0) __builtin_amdgcn_s_wait_tensorcnt(0);
    __syncthreads();                               // current tiles visible to WG
    if (wave == 0 && lk0 + 32 < kL) {              // DMA next chunk during compute
      tdm_load_to_lds((unsigned)(uintptr_t)&ldsK[cur ^ 1][0],
                      kb + (size_t)(lk0 + 32) * kD, 512, 1, 512, 1, 512);
      tdm_load_to_lds((unsigned)(uintptr_t)&ldsV[cur ^ 1][0],
                      vb + lk0 + 32, 512, 64, 8, 64, 512);
    }
#else
    {                                              // K chunk: 2048 halves, flat
      const v8h* src = (const v8h*)(kb + (size_t)lk0 * kD);
      v8h* dst = (v8h*)&ldsK[cur][0];
      dst[tid]       = src[tid];
      dst[tid + 128] = src[tid + 128];
    }
    {                                              // V chunk: 64 rows x 32 halves
      const int d  = tid >> 1;
      const int hh = (tid & 1) * 16;
      *(v16h*)(&ldsV[cur][0] + d * 32 + hh) =
          *(const v16h*)(vb + (size_t)d * kL + lk0 + hh);
    }
    __syncthreads();
#endif
    const _Float16* Kc = &ldsK[cur][0];
    const _Float16* Vc = &ldsV[cur][0];

#pragma unroll
    for (int mi = 0; mi < 2; ++mi) {
      v8f s[2] = {};
#pragma unroll
      for (int nt = 0; nt < 2; ++nt) {
#pragma unroll
        for (int kc = 0; kc < 2; ++kc) {
          const _Float16* bp =
              Kc + (nt * 16 + (lane & 15)) * 64 + kc * 32 + ((lane >> 4) << 4);
          v16h bf = *(const v16h*)bp;
          s[nt] = wmma_f16(aq[mi][kc], bf, s[nt]);
        }
      }
      float p0[8], p1[8], alpha[8];
#pragma unroll
      for (int r = 0; r < 8; ++r) {
        float s0 = s[0][r] * kScale, s1 = s[1][r] * kScale;
        float mx = fmaxf(s0, s1);
#pragma unroll
        for (int m = 1; m < 16; m <<= 1) mx = fmaxf(mx, __shfl_xor(mx, m, 16));
        float newM = fmaxf(rowM[mi][r], mx);
        alpha[r]   = __expf(rowM[mi][r] - newM);
        rowM[mi][r] = newM;
        p0[r] = __expf(s0 - newM);
        p1[r] = __expf(s1 - newM);
        float cs = p0[r] + p1[r];
#pragma unroll
        for (int m = 1; m < 16; m <<= 1) cs += __shfl_xor(cs, m, 16);
        rowL[mi][r] = rowL[mi][r] * alpha[r] + cs;
      }
#pragma unroll
      for (int j = 0; j < 4; ++j)
#pragma unroll
        for (int r = 0; r < 8; ++r) acc[mi][j][r] *= alpha[r];
      // C-layout -> row-major P[32][32] (this wave's region)
#pragma unroll
      for (int r = 0; r < 8; ++r) {
        const int row = mi * 16 + r + ((lane >> 4) << 3);
        const int col = lane & 15;
        lp[row * 32 + col]      = (_Float16)p0[r];
        lp[row * 32 + 16 + col] = (_Float16)p1[r];
      }
    }
    __syncthreads();                               // P visible across lanes
#pragma unroll
    for (int mi = 0; mi < 2; ++mi) {
      v16h ap = load_a16(lp + (mi * 16 + (lane & 15)) * 32, 0, lane);
#pragma unroll
      for (int j = 0; j < 4; ++j) {
        const _Float16* bp = Vc + (j * 16 + (lane & 15)) * 32 + ((lane >> 4) << 4);
        v16h bf = *(const v16h*)bp;
        acc[mi][j] = wmma_f16(ap, bf, acc[mi][j]);
      }
    }
    __syncthreads();                               // all reads of cur buffers done
  }
  // normalize rows, write O transposed [B][L][C] (K-contiguous for out GEMM)
#pragma unroll
  for (int mi = 0; mi < 2; ++mi)
#pragma unroll
    for (int j = 0; j < 4; ++j)
#pragma unroll
      for (int r = 0; r < 8; ++r) {
        const int lq = lq0 + mi * 16 + r + ((lane >> 4) << 3);
        const int d  = j * 16 + (lane & 15);
        oT[((size_t)(b * kL + lq)) * kC + h * kD + d] =
            (_Float16)(acc[mi][j][r] / rowL[mi][r]);
      }
}

// ---------------------------------------------------------------------------
// Kernel 6: output projection [C x C] * [C x L] + bias + residual (f32 out)
// ---------------------------------------------------------------------------
__global__ __launch_bounds__(128)
void out_gemm_kernel(const _Float16* __restrict__ wo,
                     const _Float16* __restrict__ oT,
                     const float* __restrict__ out_b,
                     const float* __restrict__ x,
                     float* __restrict__ y) {
  const int lane = threadIdx.x & 31;
  const int wave = threadIdx.x >> 5;
  const int b  = blockIdx.z;
  const int m0 = blockIdx.x * 64 + (wave & 1) * 32;
  const int n0 = blockIdx.y * 128 + (wave >> 1) * 64;
  const _Float16* Bb = oT + (size_t)b * kL * kC;

  v8f acc[2][4] = {};
  for (int kc = 0; kc < kC; kc += 32) {
    v16h a[2];
    a[0] = load_a16(wo + (size_t)(m0 + (lane & 15)) * kC, kc, lane);
    a[1] = load_a16(wo + (size_t)(m0 + 16 + (lane & 15)) * kC, kc, lane);
#pragma unroll
    for (int nj = 0; nj < 4; ++nj) {
      const _Float16* bp =
          Bb + (size_t)(n0 + nj * 16 + (lane & 15)) * kC + kc + ((lane >> 4) << 4);
      v16h bf = *(const v16h*)bp;
      acc[0][nj] = wmma_f16(a[0], bf, acc[0][nj]);
      acc[1][nj] = wmma_f16(a[1], bf, acc[1][nj]);
    }
  }
#pragma unroll
  for (int mi = 0; mi < 2; ++mi) {
#pragma unroll
    for (int r = 0; r < 8; ++r) {
      const int o = m0 + mi * 16 + r + ((lane >> 4) << 3);
      const float bias = out_b[o];
      const size_t rowbase = ((size_t)(b * kC) + o) * kL;
#pragma unroll
      for (int nj = 0; nj < 4; ++nj) {
        const int l = n0 + nj * 16 + (lane & 15);
        y[rowbase + l] = acc[mi][nj][r] + bias + x[rowbase + l];
      }
    }
  }
}

// ---------------------------------------------------------------------------
// Host launcher
// ---------------------------------------------------------------------------
extern "C" void kernel_launch(void* const* d_in, const int* in_sizes, int n_in,
                              void* d_out, int out_size, void* d_ws, size_t ws_size,
                              hipStream_t stream) {
  const float* x     = (const float*)d_in[0];
  const float* gn_w  = (const float*)d_in[1];
  const float* gn_b  = (const float*)d_in[2];
  const float* qkv_w = (const float*)d_in[3];
  const float* qkv_b = (const float*)d_in[4];
  const float* out_w = (const float*)d_in[5];
  const float* out_b = (const float*)d_in[6];
  float* y = (float*)d_out;

  // workspace carve-up (256B aligned), ~44.0 MB total
  char* ws = (char*)d_ws;
  auto carve = [&](size_t bytes) -> void* {
    void* p = (void*)ws;
    ws += (bytes + 255) & ~(size_t)255;
    return p;
  };
  float*    stats = (float*)   carve((size_t)kB * kG * 2 * sizeof(float));
  _Float16* wq    = (_Float16*)carve((size_t)3 * kC * kC * 2);
  _Float16* wo    = (_Float16*)carve((size_t)kC * kC * 2);
  _Float16* hnT   = (_Float16*)carve((size_t)kB * kL * kC * 2);
  _Float16* qh    = (_Float16*)carve((size_t)kB * kH * kL * kD * 2);
  _Float16* kh    = (_Float16*)carve((size_t)kB * kH * kL * kD * 2);
  _Float16* vT    = (_Float16*)carve((size_t)kB * kH * kD * kL * 2);
  _Float16* oT    = (_Float16*)carve((size_t)kB * kL * kC * 2);

  cast_weights_kernel<<<dim3((3 * kC * kC + 255) / 256), dim3(256), 0, stream>>>(
      qkv_w, out_w, wq, wo);
  gn_stats_kernel<<<dim3(kB * kG), dim3(256), 0, stream>>>(x, stats);
  gn_apply_kernel<<<dim3(kL / 64, kC / 32, kB), dim3(256), 0, stream>>>(
      x, gn_w, gn_b, stats, hnT);
  qkv_gemm_kernel<<<dim3(3 * kC / 64, kL / 128, kB), dim3(128), 0, stream>>>(
      wq, hnT, qkv_b, qh, kh, vT);
  attn_kernel<<<dim3(kL / 128, kH, kB), dim3(128), 0, stream>>>(qh, kh, vT, oT);
  out_gemm_kernel<<<dim3(kC / 64, kL / 128, kB), dim3(128), 0, stream>>>(
      wo, oT, out_b, x, y);
}
